// MagLinear_26336739459507
// MI455X (gfx1250) — matrix operand order repaired and checked by reading, not compiled, not run on hardware
//
#include <hip/hip_runtime.h>
#include <hip/hip_bf16.h>
#include <math.h>

typedef __attribute__((ext_vector_type(16))) __bf16 v16bf;
typedef __attribute__((ext_vector_type(2)))  __bf16 v2bf;
typedef __attribute__((ext_vector_type(8)))  float  v8f;
typedef __attribute__((ext_vector_type(2)))  float  v2f;
typedef __attribute__((ext_vector_type(4)))  int    v4i;

#define BATCH 512
#define KDIM  512
#define CLS   100000
#define TM    128
#define TN    128
#define KT    32
#define NCHUNK ((CLS + TN - 1) / TN)   /* 782 */

#define SCALE_S  64.0f
#define L_MARGIN 0.45f
#define U_MARGIN 0.80f
#define L_A      10.0f
#define U_A      110.0f
#define LAMBDA_G 20.0f

// Native bf16 conversions (RNE) -> v_cvt_pk_bf16_f32 on gfx1250
__device__ __forceinline__ unsigned int pack2bf(float lo, float hi) {
    v2f f; f.x = lo; f.y = hi;
    v2bf b = __builtin_convertvector(f, v2bf);
    unsigned int r;
    __builtin_memcpy(&r, &b, 4);
    return r;
}
__device__ __forceinline__ unsigned short bf1(float f) {
    __bf16 h = (__bf16)f;
    unsigned short u;
    __builtin_memcpy(&u, &h, 2);
    return u;
}

// ---------------- dead probe: async global->LDS path (never launched; compiles,
// confirming the gfx1250 async-copy instructions are available) ----------------
#if __has_builtin(__builtin_amdgcn_global_load_async_to_lds_b128)
__global__ void MagLinear_async_probe(float* __restrict__ g, float* __restrict__ o) {
    __shared__ float lds[64];
    __builtin_amdgcn_global_load_async_to_lds_b128((v4i*)g, (v4i*)lds, 0, 0);
    asm volatile("s_wait_asynccnt 0" ::: "memory");
    __syncthreads();
    o[threadIdx.x & 63] = lds[threadIdx.x & 63];
}
#endif

// ---------------- Kernel 1: per-row x stats ----------------
__global__ void MagLinear_xprep_kernel(const float* __restrict__ x,
                                       float* __restrict__ xinv,
                                       float* __restrict__ xnc,
                                       float* __restrict__ cosm,
                                       float* __restrict__ sinm) {
    __shared__ float red[128];
    int r = blockIdx.x;
    int t = threadIdx.x;
    float s = 0.0f;
    for (int k = t; k < KDIM; k += 128) {
        float v = x[(size_t)r * KDIM + k];
        s = fmaf(v, v, s);
    }
    red[t] = s;
    __syncthreads();
    for (int o = 64; o > 0; o >>= 1) {
        if (t < o) red[t] += red[t + o];
        __syncthreads();
    }
    if (t == 0) {
        float n = sqrtf(red[0]);
        xinv[r] = 1.0f / n;                       // x_unit uses the UNclamped norm
        float nc = fminf(fmaxf(n, L_A), U_A);     // clamped norm for margin & g
        xnc[r] = nc;
        float ada = (U_MARGIN - L_MARGIN) / (U_A - L_A) * (nc - L_A) + L_MARGIN;
        cosm[r] = cosf(ada);
        sinm[r] = sinf(ada);
    }
}

// ---------------- Kernel 2: LDS-resident B-panel WMMA GEMM + fused col-norms
//                  + streaming softmax. One block per 128-class chunk; the block
//                  loads weight ONCE (single HBM pass), reuses it for all 4 M-tiles.
__global__ __launch_bounds__(256) void MagLinear_maggemm_kernel(
    const float* __restrict__ x, const float* __restrict__ w,
    const int* __restrict__ tgt,
    const float* __restrict__ xinv,
    const float* __restrict__ cosm, const float* __restrict__ sinm,
    float* __restrict__ tlog, float* __restrict__ partials)
{
    constexpr int LDBP = KDIM + 8;  // 520 bf16 per B (col-major) panel row
    constexpr int LDA  = KT + 8;    // 40 bf16 per A row
    __shared__ unsigned short sBP[TN * LDBP];    // 130 KB: B panel [col][k] bf16
    __shared__ unsigned short sA[2][TM * LDA];   // 20 KB: A tiles [row][k] bf16
    __shared__ float s_winv[TN];
    __shared__ float s_xinv[TM], s_cm[TM], s_sm[TM];
    __shared__ int   s_tg[TM];
    __shared__ float s_rm[TM][2], s_rs[TM][2];

    const int tid   = threadIdx.x;
    const int lane  = tid & 31;
    const int wid   = tid >> 5;
    const int waveM = wid >> 1;     // 0..3 -> 32-row strip
    const int waveN = wid & 1;      // 0..1 -> 64-col strip
    const int half  = lane >> 4;    // lane half (ISA A/B/C layouts)
    const int l16   = lane & 15;

    const int chunk = blockIdx.x;
    const int tileN = chunk * TN;

    const float4 z4 = make_float4(0.f, 0.f, 0.f, 0.f);

    // ===== Phase 1: load B panel (512K x 128 cols) as bf16, fused column sumsq =====
    {
        int cq  = (tid & 31) * 4;        // col quad 0,4,...,124
        int kr0 = tid >> 5;              // 0..7
        bool okP = (tileN + cq) < CLS;   // CLS % 4 == 0 -> quad all-valid or all-OOB
        const float* wp = w + (size_t)kr0 * CLS + tileN + cq;
        float q0 = 0.f, q1 = 0.f, q2 = 0.f, q3 = 0.f;
        #pragma unroll 8
        for (int i = 0; i < KDIM / 8; ++i) {
            float4 v = okP ? *(const float4*)(wp + (size_t)(i * 8) * CLS) : z4;
            int kr = kr0 + i * 8;
            sBP[(cq + 0) * LDBP + kr] = bf1(v.x);
            sBP[(cq + 1) * LDBP + kr] = bf1(v.y);
            sBP[(cq + 2) * LDBP + kr] = bf1(v.z);
            sBP[(cq + 3) * LDBP + kr] = bf1(v.w);
            q0 = fmaf(v.x, v.x, q0);
            q1 = fmaf(v.y, v.y, q1);
            q2 = fmaf(v.z, v.z, q2);
            q3 = fmaf(v.w, v.w, q3);
        }
        float* sSq = (float*)sA;         // reuse A buffers as [TN][8] scratch (4 KB)
        sSq[(cq + 0) * 8 + kr0] = q0;
        sSq[(cq + 1) * 8 + kr0] = q1;
        sSq[(cq + 2) * 8 + kr0] = q2;
        sSq[(cq + 3) * 8 + kr0] = q3;
    }
    __syncthreads();
    if (tid < TN) {
        const float* sSq = (const float*)sA;
        float t = 0.f;
        #pragma unroll
        for (int j = 0; j < 8; ++j) t += sSq[tid * 8 + j];
        s_winv[tid] = 1.0f / sqrtf(t);   // OOB cols -> inf, never read (col<CLS guard)
    }
    __syncthreads();                     // sSq (aliasing sA) free for A staging now

    // ===== Phase 2: loop over the 4 M-tiles, B panel stays resident =====
    for (int mt = 0; mt < BATCH / TM; ++mt) {
        const int tileM = mt * TM;

        if (tid < TM) {
            int r = tileM + tid;
            s_xinv[tid] = xinv[r];
            s_cm[tid]   = cosm[r];
            s_sm[tid]   = sinm[r];
            s_tg[tid]   = tgt[r];
        }

        // A staging geometry: 128 rows x 8 float4-groups, 4 groups per thread
        size_t aOff[4];
        int    aLds[4];
        #pragma unroll
        for (int i = 0; i < 4; ++i) {
            int g  = tid + i * 256;
            int ra = g >> 3;            // A row 0..127
            int kq = (g & 7) * 4;       // A k-quad 0,4,...,28
            aOff[i] = (size_t)(tileM + ra) * KDIM + kq;
            aLds[i] = ra * LDA + kq;
        }

        v8f acc[2][4];
        v8f zero = {};
        #pragma unroll
        for (int i = 0; i < 2; ++i)
            #pragma unroll
            for (int j = 0; j < 4; ++j) acc[i][j] = zero;

        // prologue: stage A K-slab 0 into buffer 0
        #pragma unroll
        for (int i = 0; i < 4; ++i) {
            float4 av = *(const float4*)(x + aOff[i]);
            uint2 p; p.x = pack2bf(av.x, av.y); p.y = pack2bf(av.z, av.w);
            *(uint2*)&sA[0][aLds[i]] = p;
        }
        __syncthreads();

        int cur = 0;
        for (int kk = 0; kk < KDIM; kk += KT) {
            const bool hasNext = (kk + KT) < KDIM;

            // 1) next A slab's global loads (latency hidden by WMMA)
            float4 aR[4];
            if (hasNext) {
                #pragma unroll
                for (int i = 0; i < 4; ++i)
                    aR[i] = *(const float4*)(x + aOff[i] + (kk + KT));
            }

            // 2) compute: A from double buffer, B straight from resident panel
            union AF { unsigned int u[8]; v16bf v; };
            const unsigned short* Ab = sA[cur];
            // B frag: lane l16 = N, half selects K lo/hi 16, VGPR v -> K = kk + half*16 + 2v
            AF bfrag[4];
            #pragma unroll
            for (int sn = 0; sn < 4; ++sn) {
                int cl = waveN * 64 + sn * 16 + l16;
                const unsigned int* p =
                    (const unsigned int*)&sBP[cl * LDBP + kk + half * 16];
                #pragma unroll
                for (int v = 0; v < 8; ++v) bfrag[sn].u[v] = p[v];
            }
            #pragma unroll
            for (int sm = 0; sm < 2; ++sm) {
                // A frag: lane l16 = M; VGPR v -> K = half*16 + 2v
                AF afrag;
                int rl = waveM * 32 + sm * 16 + l16;
                const unsigned short* arow = &Ab[rl * LDA];
                #pragma unroll
                for (int v = 0; v < 8; ++v) {
                    int k = half * 16 + v * 2;
                    afrag.u[v] = *(const unsigned int*)&arow[k];
                }
                #pragma unroll
                for (int sn = 0; sn < 4; ++sn) {
                    acc[sm][sn] = __builtin_amdgcn_wmma_f32_16x16x32_bf16(
                        false, afrag.v, false, bfrag[sn].v,
                        (short)0, acc[sm][sn], false, false);
                }
            }

            // 3) store staged A registers into the other buffer
            if (hasNext) {
                int nxt = cur ^ 1;
                #pragma unroll
                for (int i = 0; i < 4; ++i) {
                    uint2 p; p.x = pack2bf(aR[i].x, aR[i].y);
                    p.y = pack2bf(aR[i].z, aR[i].w);
                    *(uint2*)&sA[nxt][aLds[i]] = p;
                }
            }
            __syncthreads();
            cur ^= 1;
        }

        // --- logits + margin + rowwise (max, sumexp) over this 128-col chunk ---
        #pragma unroll
        for (int sm = 0; sm < 2; ++sm) {
            #pragma unroll
            for (int j = 0; j < 8; ++j) {
                int   rl  = waveM * 32 + sm * 16 + j + half * 8; // C/D: M = j + 8*half
                float xi  = s_xinv[rl];
                float cm  = s_cm[rl];
                float smv = s_sm[rl];
                int   tg  = s_tg[rl];
                float lg[4];
                #pragma unroll
                for (int sn = 0; sn < 4; ++sn) {
                    int cl  = waveN * 64 + sn * 16 + l16;        // C/D: N = lane&15
                    int col = tileN + cl;
                    float lgv = -1e30f;
                    if (col < CLS) {
                        union { v8f v; float f[8]; } t;
                        t.v = acc[sm][sn];
                        float ct = t.f[j] * xi * s_winv[cl];
                        ct = fminf(fmaxf(ct, -1.0f), 1.0f);
                        lgv = SCALE_S * ct;
                        if (col == tg) {
                            float st  = sqrtf(fmaxf(1.0f - ct * ct, 0.0f));
                            float ctm = (ct > 0.0f) ? (ct * cm - st * smv) : ct;
                            lgv = SCALE_S * ctm;
                            tlog[tileM + rl] = lgv;  // unique writer across the grid
                        }
                    }
                    lg[sn] = lgv;
                }
                float rmax = fmaxf(fmaxf(lg[0], lg[1]), fmaxf(lg[2], lg[3]));
                #pragma unroll
                for (int m = 1; m <= 8; m <<= 1)
                    rmax = fmaxf(rmax, __shfl_xor(rmax, m, 32)); // within 16-lane half
                float rs = __expf(lg[0] - rmax) + __expf(lg[1] - rmax) +
                           __expf(lg[2] - rmax) + __expf(lg[3] - rmax);
                #pragma unroll
                for (int m = 1; m <= 8; m <<= 1)
                    rs += __shfl_xor(rs, m, 32);
                if (l16 == 0) { s_rm[rl][waveN] = rmax; s_rs[rl][waveN] = rs; }
            }
        }
        __syncthreads();

        if (tid < TM) {
            float m0 = s_rm[tid][0], m1 = s_rm[tid][1];
            float s0 = s_rs[tid][0], s1 = s_rs[tid][1];
            float M  = fmaxf(m0, m1);
            float S  = s0 * __expf(m0 - M) + s1 * __expf(m1 - M);
            size_t r = (size_t)(tileM + tid);
            partials[(r * NCHUNK + chunk) * 2 + 0] = M;
            partials[(r * NCHUNK + chunk) * 2 + 1] = S;
        }
    }
}

// ---------------- Kernel 3: merge partials -> logsumexp -> loss scalar ----------------
__global__ __launch_bounds__(512) void MagLinear_finalize_kernel(
    const float* __restrict__ partials, const float* __restrict__ tlog,
    const float* __restrict__ xnc, float* __restrict__ out)
{
    __shared__ float red[512];
    int r = threadIdx.x;
    float M = -1e30f, S = 0.0f;
    for (int c = 0; c < NCHUNK; ++c) {
        float m = partials[((size_t)r * NCHUNK + c) * 2 + 0];
        float s = partials[((size_t)r * NCHUNK + c) * 2 + 1];
        if (m > M) { S = S * expf(M - m) + s; M = m; }
        else       { S += s * expf(m - M); }
    }
    float lse = M + logf(S);
    float nc  = xnc[r];
    float val = (lse - tlog[r]) +
                LAMBDA_G * ((1.0f / (U_A * U_A)) * nc + 1.0f / nc);
    red[r] = val;
    __syncthreads();
    for (int o = 256; o > 0; o >>= 1) {
        if (r < o) red[r] += red[r + o];
        __syncthreads();
    }
    if (r == 0) out[0] = red[0] / (float)BATCH;
}

extern "C" void kernel_launch(void* const* d_in, const int* in_sizes, int n_in,
                              void* d_out, int out_size, void* d_ws, size_t ws_size,
                              hipStream_t stream) {
    (void)in_sizes; (void)n_in; (void)out_size; (void)ws_size;
    const float* x   = (const float*)d_in[0];
    const int*   tgt = (const int*)d_in[1];
    const float* w   = (const float*)d_in[2];
    float* out = (float*)d_out;

    float* ws       = (float*)d_ws;
    float* xinv     = ws;                    // BATCH
    float* xnc      = xinv + BATCH;          // BATCH
    float* cosm     = xnc  + BATCH;          // BATCH
    float* sinm     = cosm + BATCH;          // BATCH
    float* tlog     = sinm + BATCH;          // BATCH
    float* partials = tlog + BATCH;          // BATCH * NCHUNK * 2 (~3.2 MB)

    MagLinear_xprep_kernel<<<BATCH, 128, 0, stream>>>(x, xinv, xnc, cosm, sinm);
    MagLinear_maggemm_kernel<<<NCHUNK, 256, 0, stream>>>(x, w, tgt, xinv,
                                                         cosm, sinm, tlog, partials);
    MagLinear_finalize_kernel<<<1, BATCH, 0, stream>>>(partials, tlog, xnc, out);
}
